// MetaRL_LightGAT_BiACT_58299886075964
// MI455X (gfx1250) — matrix operand order repaired and checked by reading, not compiled, not run on hardware
//
#include <hip/hip_runtime.h>
#include <hip/hip_bf16.h>

typedef __attribute__((ext_vector_type(16))) _Float16 v16h;
typedef __attribute__((ext_vector_type(8)))  float    v8f;

#define NN  8192
#define DIN 128
#define DH  48

// ---- CDNA5 wave32 WMMA fragment index maps (ISA 7.12.2, 16-bit A 16x32) ----
// A: lane l -> row M = l%16 ; element t -> K = (l<16?0:8) + (t<8 ? t : 16+(t-8))
__device__ __forceinline__ int a_k(int lane, int t) {
  int base = (lane < 16) ? 0 : 8;
  return (t < 8) ? (base + t) : (base + 16 + (t - 8));
}
// B: lane l -> col N = l%16 ; element t -> K = (l<16?0:8) + (t<8 ? t : 16+(t-8))
__device__ __forceinline__ int b_k(int lane, int t) {
  int base = (lane < 16) ? 0 : 8;
  return (t < 8) ? (base + t) : (base + 16 + (t - 8));
}
// C/D: lane l -> col N = l%16 ; vgpr r -> row M = r + (l<16?0:8)

__device__ __forceinline__ v8f wmma16(v16h a, v16h b, v8f c) {
  return __builtin_amdgcn_wmma_f32_16x16x32_f16(
      /*neg_a=*/false, a, /*neg_b=*/false, b,
      /*c_mod=*/(short)0, c, /*reuse_a=*/false, /*reuse_b=*/false);
}

__device__ __forceinline__ float red16(float v) {  // sum across a 16-lane half
  v += __shfl_xor(v, 1, 32);
  v += __shfl_xor(v, 2, 32);
  v += __shfl_xor(v, 4, 32);
  v += __shfl_xor(v, 8, 32);
  return v;
}

// ---------------------------------------------------------------------------
// Kernel 1: Wh = x @ Wgat^T  (8192x128 @ 128x48), store WhT as f16 [48][8192]
//           and s = Wh @ a^T  (attention scores, f32 [8192])
// ---------------------------------------------------------------------------
__global__ void __launch_bounds__(32)
k_gat_wh(const float* __restrict__ x, const float* __restrict__ Wg,
         const float* __restrict__ avec, _Float16* __restrict__ WhT,
         float* __restrict__ s) {
  const int lane = threadIdx.x;
  const int i0   = blockIdx.x * 16;
  const int m    = lane & 15;
  const int half = lane >> 4;

  v8f C[3];
#pragma unroll
  for (int c = 0; c < 3; ++c)
#pragma unroll
    for (int r = 0; r < 8; ++r) C[c][r] = 0.f;

  const float* xrow = x + (size_t)(i0 + m) * DIN;
#pragma unroll
  for (int kc = 0; kc < 4; ++kc) {           // K = 128 in 4 chunks of 32
    v16h A;
#pragma unroll
    for (int t = 0; t < 16; ++t)
      A[t] = (_Float16)xrow[kc * 32 + a_k(lane, t)];
#pragma unroll
    for (int c = 0; c < 3; ++c) {            // 48 output cols = 3 N-tiles
      const float* wrow = Wg + (size_t)(c * 16 + m) * DIN + kc * 32;
      v16h B;
#pragma unroll
      for (int t = 0; t < 16; ++t)
        B[t] = (_Float16)wrow[b_k(lane, t)];
      C[c] = wmma16(A, B, C[c]);
    }
  }

#pragma unroll
  for (int r = 0; r < 8; ++r) {
    const int grow = i0 + r + half * 8;
    float part = 0.f;
#pragma unroll
    for (int c = 0; c < 3; ++c) {
      float v = C[c][r];
      WhT[(size_t)(c * 16 + m) * NN + grow] = (_Float16)v;   // transposed f16
      part += v * avec[c * 16 + m];
    }
    part = red16(part);
    if (m == 0) s[grow] = part;
  }
}

// ---------------------------------------------------------------------------
// Kernel 2: fused masked-softmax attention + h' = attn @ Wh + LayerNorm.
// One wave per 16-row tile; streams adj exactly once (268 MB -> HBM-bound).
// Barrier-free, software-pipelined: adj/s for chunk j+1 are loaded straight
// into per-lane registers (4x int4 / 4x float4 covering the fragment's
// contiguous K-runs) while chunk j's exp + WMMA work executes.
// ---------------------------------------------------------------------------
struct Chunk {
  int4   q0, q1, q2, q3;    // adj: K = kb+0..3, kb+4..7, kb+8..11, kb+12..15
  float4 f0, f1, f2, f3;    // s_j at the same K positions
};

__device__ __forceinline__ Chunk load_chunk(const int* __restrict__ adjRow,
                                            const float* __restrict__ s,
                                            int j0, int kb) {
  Chunk c;
  c.q0 = *(const int4*)  (adjRow + j0 + kb);
  c.q1 = *(const int4*)  (adjRow + j0 + kb + 4);
  c.q2 = *(const int4*)  (adjRow + j0 + kb + 8);
  c.q3 = *(const int4*)  (adjRow + j0 + kb + 12);
  c.f0 = *(const float4*)(s      + j0 + kb);
  c.f1 = *(const float4*)(s      + j0 + kb + 4);
  c.f2 = *(const float4*)(s      + j0 + kb + 8);
  c.f3 = *(const float4*)(s      + j0 + kb + 12);
  return c;
}

__device__ __forceinline__ v16h make_A(const Chunk& c, float si) {
  const float sv[16] = {c.f0.x, c.f0.y, c.f0.z, c.f0.w,
                        c.f1.x, c.f1.y, c.f1.z, c.f1.w,
                        c.f2.x, c.f2.y, c.f2.z, c.f2.w,
                        c.f3.x, c.f3.y, c.f3.z, c.f3.w};
  const int   av[16] = {c.q0.x, c.q0.y, c.q0.z, c.q0.w,
                        c.q1.x, c.q1.y, c.q1.z, c.q1.w,
                        c.q2.x, c.q2.y, c.q2.z, c.q2.w,
                        c.q3.x, c.q3.y, c.q3.z, c.q3.w};
  v16h A;
#pragma unroll
  for (int t = 0; t < 16; ++t) {
    float e = si + sv[t];
    e = fmaxf(e, 0.2f * e);                         // LeakyReLU(0.2)
    A[t] = (av[t] > 0) ? (_Float16)__expf(e) : (_Float16)0.f;
  }
  return A;
}

__global__ void __launch_bounds__(32)
k_gat_attn(const int* __restrict__ adj, const float* __restrict__ s,
           const _Float16* __restrict__ WhT, const float* __restrict__ gamma,
           const float* __restrict__ beta, float* __restrict__ h) {
  const int lane = threadIdx.x;
  const int i0   = blockIdx.x * 16;
  const int m    = lane & 15;
  const int half = lane >> 4;
  const int kb   = half * 16;                 // per-lane K-run base

  const float si = s[i0 + m];                 // score of this lane's A-row
  const int* adjRow = adj + (size_t)(i0 + m) * NN;

  v8f C[3], D;
#pragma unroll
  for (int c = 0; c < 3; ++c)
#pragma unroll
    for (int r = 0; r < 8; ++r) C[c][r] = 0.f;
#pragma unroll
  for (int r = 0; r < 8; ++r) D[r] = 0.f;

  v16h Bone;
#pragma unroll
  for (int t = 0; t < 16; ++t) Bone[t] = (_Float16)1.f;  // denominator B tile

  Chunk cur = load_chunk(adjRow, s, 0, kb);   // prologue: chunk 0

  for (int j0 = 0; j0 < NN; j0 += 32) {
    // --- issue next chunk's loads first (pipelined past the exp/WMMA) ---
    const int jn = (j0 + 32 < NN) ? j0 + 32 : j0;
    Chunk nxt = load_chunk(adjRow, s, jn, kb);
    if (j0 + 64 < NN)                         // warm L2 two chunks ahead
      __builtin_prefetch(adjRow + j0 + 64 + kb, 0, 0);

    // --- numerator tile p = exp(leakyrelu(s_i+s_j)) masked by adj ---
    const v16h A = make_A(cur, si);

#pragma unroll
    for (int c = 0; c < 3; ++c) {
      const v16h B = *(const v16h*)(WhT + (size_t)(c * 16 + m) * NN + j0 + half * 16);
      C[c] = wmma16(A, B, C[c]);
    }
    D = wmma16(A, Bone, D);                   // row sums (denominator)

    cur = nxt;
  }

  // --- normalize by denominator, then LayerNorm over the 48 features ---
#pragma unroll
  for (int r = 0; r < 8; ++r) {
    const int grow = i0 + r + half * 8;
    const float rdenom = 1.f / fmaxf(D[r], 1e-20f);
    const float v0 = C[0][r] * rdenom;
    const float v1 = C[1][r] * rdenom;
    const float v2 = C[2][r] * rdenom;
    const float sum = red16(v0 + v1 + v2);
    const float sq  = red16(v0 * v0 + v1 * v1 + v2 * v2);
    const float mean = sum * (1.f / 48.f);
    const float var  = sq * (1.f / 48.f) - mean * mean;
    const float rs   = rsqrtf(var + 1e-5f);
    float* hrow = h + (size_t)grow * DH;
    hrow[m]      = (v0 - mean) * rs * gamma[m]      + beta[m];
    hrow[16 + m] = (v1 - mean) * rs * gamma[16 + m] + beta[16 + m];
    hrow[32 + m] = (v2 - mean) * rs * gamma[32 + m] + beta[32 + m];
  }
}

// ---------------------------------------------------------------------------
// Kernel 3: MLP head 48 -> 256 -> 128 -> 32 (relu, relu, linear), all WMMA.
// LDS round-trips provide the C-layout -> A-layout transpose between layers.
// ---------------------------------------------------------------------------
__global__ void __launch_bounds__(32)
k_mlp(const float* __restrict__ h,
      const float* __restrict__ W1, const float* __restrict__ b1,
      const float* __restrict__ W2, const float* __restrict__ b2,
      const float* __restrict__ W3, const float* __restrict__ b3,
      float* __restrict__ out) {
  __shared__ __align__(16) _Float16 lds[16 * 256];

  const int lane = threadIdx.x;
  const int i0   = blockIdx.x * 16;
  const int m    = lane & 15;
  const int half = lane >> 4;

  // ---- Layer 1: K=48 (chunk of 32 + zero-padded chunk of 16) ----
  v16h A0, A1;
  {
    const float* hrow = h + (size_t)(i0 + m) * DH;
#pragma unroll
    for (int t = 0; t < 16; ++t) A0[t] = (_Float16)hrow[a_k(lane, t)];
#pragma unroll
    for (int t = 0; t < 16; ++t) {
      const int k = 32 + a_k(lane, t);
      A1[t] = (k < DH) ? (_Float16)hrow[k] : (_Float16)0.f;
    }
  }
#pragma unroll
  for (int nt = 0; nt < 16; ++nt) {          // 256 cols = 16 N-tiles
    const int col = nt * 16 + m;
    const float bias = b1[col];
    v8f C;
#pragma unroll
    for (int r = 0; r < 8; ++r) C[r] = bias;
    const float* w = W1 + (size_t)col * DH;
    v16h B0, B1;
#pragma unroll
    for (int t = 0; t < 16; ++t) B0[t] = (_Float16)w[b_k(lane, t)];
#pragma unroll
    for (int t = 0; t < 16; ++t) {
      const int k = 32 + b_k(lane, t);
      B1[t] = (k < DH) ? (_Float16)w[k] : (_Float16)0.f;
    }
    C = wmma16(A0, B0, C);
    C = wmma16(A1, B1, C);
#pragma unroll
    for (int r = 0; r < 8; ++r)
      lds[(r + half * 8) * 256 + col] = (_Float16)fmaxf(C[r], 0.f);  // ReLU
  }
  __syncthreads();

  // ---- Layer 2: K=256, prebuild all A fragments, then reuse LDS ----
  v16h A2[8];
#pragma unroll
  for (int kc = 0; kc < 8; ++kc)
#pragma unroll
    for (int t = 0; t < 16; ++t)
      A2[kc][t] = lds[m * 256 + kc * 32 + a_k(lane, t)];
  __syncthreads();                            // all reads done before overwrite

#pragma unroll
  for (int nt = 0; nt < 8; ++nt) {            // 128 cols = 8 N-tiles
    const int col = nt * 16 + m;
    const float bias = b2[col];
    v8f C;
#pragma unroll
    for (int r = 0; r < 8; ++r) C[r] = bias;
    const float* w = W2 + (size_t)col * 256;
#pragma unroll
    for (int kc = 0; kc < 8; ++kc) {
      v16h B;
#pragma unroll
      for (int t = 0; t < 16; ++t) B[t] = (_Float16)w[kc * 32 + b_k(lane, t)];
      C = wmma16(A2[kc], B, C);
    }
#pragma unroll
    for (int r = 0; r < 8; ++r)
      lds[(r + half * 8) * 128 + col] = (_Float16)fmaxf(C[r], 0.f);  // ReLU
  }
  __syncthreads();

  // ---- Layer 3: K=128 -> 32 outputs ----
  v16h A3[4];
#pragma unroll
  for (int kc = 0; kc < 4; ++kc)
#pragma unroll
    for (int t = 0; t < 16; ++t)
      A3[kc][t] = lds[m * 128 + kc * 32 + a_k(lane, t)];

#pragma unroll
  for (int nt = 0; nt < 2; ++nt) {            // 32 cols = 2 N-tiles
    const int col = nt * 16 + m;
    const float bias = b3[col];
    v8f C;
#pragma unroll
    for (int r = 0; r < 8; ++r) C[r] = bias;
    const float* w = W3 + (size_t)col * 128;
#pragma unroll
    for (int kc = 0; kc < 4; ++kc) {
      v16h B;
#pragma unroll
      for (int t = 0; t < 16; ++t) B[t] = (_Float16)w[kc * 32 + b_k(lane, t)];
      C = wmma16(A3[kc], B, C);
    }
#pragma unroll
    for (int r = 0; r < 8; ++r)
      out[(size_t)(i0 + r + half * 8) * 32 + col] = C[r];
  }
}

// ---------------------------------------------------------------------------
extern "C" void kernel_launch(void* const* d_in, const int* in_sizes, int n_in,
                              void* d_out, int out_size, void* d_ws, size_t ws_size,
                              hipStream_t stream) {
  const float* x     = (const float*)d_in[0];
  const int*   adj   = (const int*)  d_in[1];
  const float* Wg    = (const float*)d_in[2];
  const float* avec  = (const float*)d_in[3];
  const float* gamma = (const float*)d_in[4];
  const float* beta  = (const float*)d_in[5];
  const float* W1    = (const float*)d_in[6];
  const float* b1    = (const float*)d_in[7];
  const float* W2    = (const float*)d_in[8];
  const float* b2    = (const float*)d_in[9];
  const float* W3    = (const float*)d_in[10];
  const float* b3    = (const float*)d_in[11];

  char* ws = (char*)d_ws;
  _Float16* WhT = (_Float16*)ws;                        // 48*8192*2  = 786432 B
  float*    s   = (float*)(ws + 786432);                // 8192*4     =  32768 B
  float*    h   = (float*)(ws + 786432 + 32768);        // 8192*48*4  = 1572864 B

  dim3 grid(NN / 16), block(32);
  k_gat_wh  <<<grid, block, 0, stream>>>(x, Wg, avec, WhT, s);
  k_gat_attn<<<grid, block, 0, stream>>>(adj, s, WhT, gamma, beta, h);
  k_mlp     <<<grid, block, 0, stream>>>(h, W1, b1, W2, b2, W3, b3, (float*)d_out);
}
